// DotPructAttention_45724221833893
// MI455X (gfx1250) — matrix operand order repaired and checked by reading, not compiled, not run on hardware
//
#include <hip/hip_runtime.h>
#include <hip/hip_bf16.h>

// ---------------------------------------------------------------------------
// Flash-attention for MI455X (gfx1250, wave32, WMMA bf16), software-pipelined,
// lean-VALU inner loop: pk-f32 vector ops, asm v_max_num, packed P stores,
// deferred l-reduction, hoisted masking.
// Shapes fixed by the reference: BH=128, Q_LEN=K_LEN=1024, HEAD_DIM=64.
// ---------------------------------------------------------------------------

typedef __bf16 bf16_t;
typedef __attribute__((ext_vector_type(16))) __bf16 v16bf;
typedef __attribute__((ext_vector_type(8)))  float  v8f;

#define BH     128
#define HEADS  16
#define QL     1024
#define KLEN   1024
#define HD     64
#define KT     32          // keys processed per iteration
#define WAVES  8
#define QBLOCK 128         // query rows per workgroup (16 per wave)
#define L2E    1.44269504088896340736f

// A-matrix (16x32 bf16) per-lane K index for element e, lane-half h.
__device__ __forceinline__ int ka_map(int e, int h) {
    int p = e >> 1;
    int base = (p < 4) ? (2 * p) : (2 * p + 8);
    return base + 8 * h + (e & 1);
}

__device__ __forceinline__ v8f wmma_bf16(v16bf a, v16bf b, v8f c) {
    return __builtin_amdgcn_wmma_f32_16x16x32_bf16(
        false, a, false, b, (short)0, c, false, false);
}

// Single v_max_num_f32 (no canonicalize pair, no med3 expansion).
__device__ __forceinline__ float fmax2(float a, float b) {
    float d;
    asm("v_max_num_f32 %0, %1, %2" : "=v"(d) : "v"(a), "v"(b));
    return d;
}

// xor-lane permutation within each 16-lane row via v_permlane16_b32 (VALU,
// co-executes with WMMA; keeps the LDS pipe free for tile traffic).
__device__ __forceinline__ float permx(float v, unsigned lo, unsigned hi) {
    unsigned u = __builtin_bit_cast(unsigned, v);
    u = __builtin_amdgcn_permlane16(u, u, lo, hi, false, false);
    return __builtin_bit_cast(float, u);
}
__device__ __forceinline__ float rowmax16(float t) {
    t = fmax2(t, permx(t, 0x67452301u, 0xEFCDAB89u));   // xor 1
    t = fmax2(t, permx(t, 0x54761032u, 0xDCFE98BAu));   // xor 2
    t = fmax2(t, permx(t, 0x32107654u, 0xBA98FEDCu));   // xor 4
    t = fmax2(t, permx(t, 0xFEDCBA98u, 0x76543210u));   // xor 8
    return t;
}
__device__ __forceinline__ float rowsum16(float t) {
    t += permx(t, 0x67452301u, 0xEFCDAB89u);
    t += permx(t, 0x54761032u, 0xDCFE98BAu);
    t += permx(t, 0x32107654u, 0xBA98FEDCu);
    t += permx(t, 0xFEDCBA98u, 0x76543210u);
    return t;
}

__global__ __launch_bounds__(256, 1)
void fa_cdna5_kernel(const float* __restrict__ Q,
                     const float* __restrict__ K,
                     const float* __restrict__ V,
                     const int*   __restrict__ vlens,
                     float*       __restrict__ O)
{
    // Double-buffered LDS tiles + per-wave packed-P scratch (C->A bounce).
    __shared__ bf16_t   Ks[2][KT][HD + 8];   // [buf][key][d], row 144B
    __shared__ bf16_t   Vt[2][HD][KT + 8];   // [buf][d][key], row 80B
    __shared__ unsigned PwU[WAVES][16][18];  // packed bf16 pairs, row 72B

    const int tid  = threadIdx.x;
    const int lane = tid & 31;
    const int wave = tid >> 5;
    const int h    = lane >> 4;
    const int ln   = lane & 15;

    const int wg    = blockIdx.x;
    const int bh    = wg >> 3;
    const int qblk  = wg & 7;
    const int qbase = qblk * QBLOCK + wave * 16;
    const int vlen  = vlens[bh / HEADS];

    const float* Qp = Q + (size_t)bh * QL * HD;
    const float* Kp = K + (size_t)bh * KLEN * HD;
    const float* Vp = V + (size_t)bh * KLEN * HD;
    float*       Op = O + (size_t)bh * QL * HD;

    // Cooperative tile staging: 32x64 fp32 = 512 float4 per tensor.
    const int rowA = tid >> 4,         colA = (tid & 15) << 2;
    const int rowB = (tid + 256) >> 4, colB = (tid & 15) << 2;

    float4 kr0, kr1, vr0, vr1;
    auto load_tile = [&](int kk0) {
        kr0 = *(const float4*)(Kp + (size_t)(kk0 + rowA) * HD + colA);
        kr1 = *(const float4*)(Kp + (size_t)(kk0 + rowB) * HD + colB);
        vr0 = *(const float4*)(Vp + (size_t)(kk0 + rowA) * HD + colA);
        vr1 = *(const float4*)(Vp + (size_t)(kk0 + rowB) * HD + colB);
    };
    auto store_tile = [&](int bi) {
        Ks[bi][rowA][colA + 0] = (bf16_t)kr0.x;
        Ks[bi][rowA][colA + 1] = (bf16_t)kr0.y;
        Ks[bi][rowA][colA + 2] = (bf16_t)kr0.z;
        Ks[bi][rowA][colA + 3] = (bf16_t)kr0.w;
        Ks[bi][rowB][colB + 0] = (bf16_t)kr1.x;
        Ks[bi][rowB][colB + 1] = (bf16_t)kr1.y;
        Ks[bi][rowB][colB + 2] = (bf16_t)kr1.z;
        Ks[bi][rowB][colB + 3] = (bf16_t)kr1.w;
        Vt[bi][colA + 0][rowA] = (bf16_t)vr0.x;
        Vt[bi][colA + 1][rowA] = (bf16_t)vr0.y;
        Vt[bi][colA + 2][rowA] = (bf16_t)vr0.z;
        Vt[bi][colA + 3][rowA] = (bf16_t)vr0.w;
        Vt[bi][colB + 0][rowB] = (bf16_t)vr1.x;
        Vt[bi][colB + 1][rowB] = (bf16_t)vr1.y;
        Vt[bi][colB + 2][rowB] = (bf16_t)vr1.z;
        Vt[bi][colB + 3][rowB] = (bf16_t)vr1.w;
    };
    auto prefetch_tile = [&](int kk0) {       // global_prefetch_b8, 64B/lane
        if (tid < 128) {
            const int row = tid >> 2;
            const int c   = (tid & 3) << 4;
            __builtin_prefetch(Kp + (size_t)(kk0 + row) * HD + c, 0, 1);
        } else {
            const int t   = tid - 128;
            const int row = t >> 2;
            const int c   = (t & 3) << 4;
            __builtin_prefetch(Vp + (size_t)(kk0 + row) * HD + c, 0, 1);
        }
    };

    // ---- Q A-fragments (16 rows x 64 d), pre-scaled by 1/sqrt(64) ----
    v16bf qa0, qa1;
    {
        const float* qrow = Qp + (size_t)(qbase + ln) * HD;
#pragma unroll
        for (int e = 0; e < 16; ++e) {
            const int k = ka_map(e, h);
            qa0[e] = (bf16_t)(qrow[k]      * 0.125f);
            qa1[e] = (bf16_t)(qrow[32 + k] * 0.125f);
        }
    }

    const v8f zero8 = {0.f, 0.f, 0.f, 0.f, 0.f, 0.f, 0.f, 0.f};
    v8f o[4];
#pragma unroll
    for (int f = 0; f < 4; ++f) o[f] = zero8;

    float m[8];            // row max (uniform across the 16-lane row)
    v8f   lv = zero8;      // PER-LANE partial softmax denominator
#pragma unroll
    for (int i = 0; i < 8; ++i) m[i] = -3.0e38f;

    // Tiles past valid_len have exactly-zero softmax weight (exp underflow);
    // valid_len==0 must process everything (uniform softmax over -1e6).
    int iters = (vlen == 0) ? (KLEN / KT) : ((vlen + KT - 1) / KT);
    if (iters > KLEN / KT) iters = KLEN / KT;

    // ---- Pipeline prologue: tile 0 -> buf 0 ----
    load_tile(0);
    store_tile(0);
    __syncthreads();

    for (int it = 0; it < iters; ++it) {
        const int kk0 = it * KT;
        const int cur = it & 1;
        const int nxt = cur ^ 1;

        // Issue next tile's global loads early (latency hidden under WMMAs).
        if (it + 1 < iters) load_tile(kk0 + KT);
        if (it + 2 < iters) prefetch_tile(kk0 + 2 * KT);

        // ---- S = Q * K^T (contract d=64; two 16x16 score tiles) ----
        v8f s0 = zero8, s1 = zero8;
#pragma unroll
        for (int f = 0; f < 2; ++f) {
            v16bf kb0, kb1;   // B 32x16: col n = ln, row kd = ka_map(e, h)
#pragma unroll
            for (int e = 0; e < 16; ++e) {
                const int d = 32 * f + ka_map(e, h);
                kb0[e] = Ks[cur][ln][d];
                kb1[e] = Ks[cur][ln + 16][d];
            }
            s0 = wmma_bf16(f == 0 ? qa0 : qa1, kb0, s0);
            s1 = wmma_bf16(f == 0 ? qa0 : qa1, kb1, s1);
        }

        // ---- Masking: uniform branch, only straddling / fully-masked tiles
        if (kk0 + KT > vlen) {
#pragma unroll
            for (int i = 0; i < 8; ++i) {
                if (kk0 + ln      >= vlen) s0[i] = -1.0e6f;
                if (kk0 + 16 + ln >= vlen) s1[i] = -1.0e6f;
            }
        }

        // ---- Online softmax (C layout: VGPR i -> row i + 8h) ----
        v8f av, mnv;
#pragma unroll
        for (int i = 0; i < 8; ++i) {
            const float t  = rowmax16(fmax2(s0[i], s1[i]));
            const float mo = m[i];
            const float mn = fmax2(mo, t);
            m[i]   = mn;
            mnv[i] = mn;
            av[i]  = __builtin_amdgcn_exp2f((mo - mn) * L2E);
        }
        // pk-f32 vector math for exp arguments; exp itself is scalar TRANS.
        const v8f a0 = (s0 - mnv) * L2E;
        const v8f a1 = (s1 - mnv) * L2E;
        v8f p0v, p1v;
#pragma unroll
        for (int i = 0; i < 8; ++i) {
            p0v[i] = __builtin_amdgcn_exp2f(a0[i]);
            p1v[i] = __builtin_amdgcn_exp2f(a1[i]);
        }
        lv = lv * av + (p0v + p1v);   // pk fma + pk add (per-lane partial)

        // ---- Packed C-layout P -> LDS (one b32 per row: bf16 pair) ----
        // Layout: key k lives at u16 slot 2*(k&15) + (k>>4) of row (q row).
#pragma unroll
        for (int i = 0; i < 8; ++i) {
            const unsigned pk =
                  (unsigned)__builtin_bit_cast(unsigned short, (bf16_t)p0v[i])
                | ((unsigned)__builtin_bit_cast(unsigned short, (bf16_t)p1v[i]) << 16);
            PwU[wave][i + 8 * h][ln] = pk;
        }
        asm volatile("s_wait_dscnt 0" ::: "memory");
        __builtin_amdgcn_wave_barrier();

        v16bf pa;   // A 16x32: lane row M = ln, K = ka_map(e, h)
        {
            const bf16_t* prow = (const bf16_t*)&PwU[wave][ln][0];
#pragma unroll
            for (int e = 0; e < 16; ++e) {
                const int k = ka_map(e, h);
                pa[e] = prow[((k & 15) << 1) | (k >> 4)];
            }
        }

        // ---- Rescale O (pk muls), then O += P * V ----
#pragma unroll
        for (int f = 0; f < 4; ++f) o[f] = o[f] * av;

#pragma unroll
        for (int f = 0; f < 4; ++f) {
            v16bf vb;   // B 32x16: col n -> d = 16f + ln, row = key ka_map(e, h)
#pragma unroll
            for (int e = 0; e < 16; ++e)
                vb[e] = Vt[cur][16 * f + ln][ka_map(e, h)];
            o[f] = wmma_bf16(pa, vb, o[f]);
        }

        // ---- Stage next tile into the alternate buffer, single barrier ----
        if (it + 1 < iters) store_tile(nxt);
        __syncthreads();
    }

    // ---- Epilogue: reduce per-lane l partials, normalize, store fp32 ----
    float rl[8];
#pragma unroll
    for (int i = 0; i < 8; ++i) rl[i] = 1.0f / rowsum16(lv[i]);

#pragma unroll
    for (int f = 0; f < 4; ++f) {
#pragma unroll
        for (int i = 0; i < 8; ++i) {
            const int row = qbase + i + 8 * h;
            const int d   = 16 * f + ln;
            Op[(size_t)row * HD + d] = o[f][i] * rl[i];
        }
    }
}

extern "C" void kernel_launch(void* const* d_in, const int* in_sizes, int n_in,
                              void* d_out, int out_size, void* d_ws, size_t ws_size,
                              hipStream_t stream) {
    const float* Q  = (const float*)d_in[0];
    const float* K  = (const float*)d_in[1];
    const float* V  = (const float*)d_in[2];
    const int*   vl = (const int*)d_in[3];
    float*       O  = (float*)d_out;

    dim3 grid(BH * (QL / QBLOCK));   // 1024 workgroups
    dim3 block(256);                 // 8 waves (wave32)
    fa_cdna5_kernel<<<grid, block, 0, stream>>>(Q, K, V, vl, O);
}
